// TransformerEncoder_36137854828976
// MI455X (gfx1250) — matrix operand order repaired and checked by reading, not compiled, not run on hardware
//
#include <hip/hip_runtime.h>
#include <hip/hip_bf16.h>

#define B_  2
#define S_  2048
#define D_  1024
#define H_  16
#define HD_ 64
#define F_  4096
#define L_  8
#define M_  (B_*S_)
#define EPS_ 1e-5f

typedef __attribute__((ext_vector_type(16))) __bf16 bf16x16;
typedef __attribute__((ext_vector_type(8)))  float  floatx8;

#define FLAG_RES  1
#define FLAG_GELU 2
#define FLAG_QKV  4

// Probe for gfx1250 async LDS copy builtins (guarded: falls back if absent)
#if defined(__has_builtin)
#  if __has_builtin(__builtin_amdgcn_global_load_async_to_lds_b128) && \
      __has_builtin(__builtin_amdgcn_global_store_async_from_lds_b128) && \
      __has_builtin(__builtin_amdgcn_s_wait_asynccnt)
#    define USE_ASYNC_COPY 1
#  endif
#endif
#ifndef USE_ASYNC_COPY
#  define USE_ASYNC_COPY 0
#endif

#if USE_ASYNC_COPY
// builtin signature (from hipcc diagnostic): int __vector(4) pointers, global AS first
typedef __attribute__((ext_vector_type(4))) int i32x4_t;
typedef __attribute__((address_space(1))) i32x4_t g_i32x4;
typedef __attribute__((address_space(3))) i32x4_t l_i32x4;
#endif

// native bf16 convert (lowers to v_cvt_pk_bf16_f32 pairs)
static __device__ __forceinline__ __bf16 f2bf(float f) { return (__bf16)f; }

// -------------------------- embedding gather --------------------------
__global__ __launch_bounds__(256) void embed_kernel(const int* __restrict__ ids,
                                                    const float* __restrict__ tok,
                                                    const float* __restrict__ pos,
                                                    float* __restrict__ x) {
  int row = blockIdx.x;              // b*S + s
  int s   = row & (S_ - 1);
  long t  = (long)ids[row] * D_;
  float* xr = x + (long)row * D_;
  const float* pr = pos + (long)s * D_;
  for (int i = threadIdx.x; i < D_; i += 256)
    xr[i] = tok[t + i] + pr[i];
}

// -------------------------- layer norm --------------------------
__global__ __launch_bounds__(256) void ln_kernel(const float* __restrict__ x,
                                                 const float* __restrict__ g,
                                                 const float* __restrict__ b,
                                                 float* __restrict__ out) {
  int row = blockIdx.x;
  const float* xr = x + (long)row * D_;
  __shared__ float red[256];
  float s = 0.f;
  for (int i = threadIdx.x; i < D_; i += 256) s += xr[i];
  red[threadIdx.x] = s; __syncthreads();
  for (int o = 128; o > 0; o >>= 1) {
    if (threadIdx.x < o) red[threadIdx.x] += red[threadIdx.x + o];
    __syncthreads();
  }
  float mean = red[0] * (1.f / D_);
  __syncthreads();
  float v = 0.f;
  for (int i = threadIdx.x; i < D_; i += 256) { float d = xr[i] - mean; v += d * d; }
  red[threadIdx.x] = v; __syncthreads();
  for (int o = 128; o > 0; o >>= 1) {
    if (threadIdx.x < o) red[threadIdx.x] += red[threadIdx.x + o];
    __syncthreads();
  }
  float rstd = rsqrtf(red[0] * (1.f / D_) + EPS_);
  float* orow = out + (long)row * D_;
  for (int i = threadIdx.x; i < D_; i += 256)
    orow[i] = (xr[i] - mean) * rstd * g[i] + b[i];
}

// -------------------------- bf16 WMMA GEMM (double buffered) --------------------------
// C[m,n] (+= if RES) A[m,:] . W(:,n) + bias[n]   with optional GELU / QKV scatter.
// W addressing: elem(k,n) at W[(n>>hs)*sH + (n&((1<<hs)-1)) + k*sK]   (n-contiguous)
#define BM 128
#define BN 64
#define BK 32
#define LDT (BK + 8)

__global__ __launch_bounds__(256) void gemm_bf16_wmma(
    const float* __restrict__ A, int K,
    const float* __restrict__ W, long sH, long sK, int hs,
    const float* __restrict__ bias,
    float* __restrict__ C, int N, int flags)
{
  __shared__ __bf16 As[2][BM][LDT];   // 2*128*40*2 = 20480 B
  __shared__ __bf16 Bs[2][BN][LDT];   // 2* 64*40*2 = 10240 B  (Bs[n][k])

  const int tid  = threadIdx.x;
  const int lane = tid & 31;
  const int wave = tid >> 5;
  const int m0 = (wave & 3) * 32;
  const int n0 = (wave >> 2) * 32;
  const int blockM = blockIdx.y * BM;
  const int blockN = blockIdx.x * BN;

  // fragment addressing (ISA 7.12.2)
  const int am  = lane & 15;
  const int akb = (lane >> 4) * 8;       // A: K interleave 0..7 / 8..15
  const int bn  = lane & 15;
  const int bk0 = (lane >> 4) * 8;       // B: K interleave 0..7 / 8..15 (same as A)

  // A tile loader: slot i -> row (tid>>3)+i*32, cols ac4..ac4+3
  const int ac4 = (tid & 7) * 4;
  // B tile loader: coalesced along n: k = tid>>3, n = (tid&7)*8 .. +7
  const int bkk = tid >> 3;
  const int bn8 = (tid & 7) * 8;
  const long baseN = (long)(blockN >> hs) * sH + (long)(blockN & ((1 << hs) - 1));

  float4 arg[4]; float4 brg[2];
  auto loadRegs = [&](int kt) {
    #pragma unroll
    for (int i = 0; i < 4; i++) {
      int row = (tid >> 3) + i * 32;
      arg[i] = *(const float4*)(A + (long)(blockM + row) * K + kt + ac4);
    }
    const float* bp = W + baseN + (long)(kt + bkk) * sK + bn8;
    brg[0] = *(const float4*)bp;
    brg[1] = *(const float4*)(bp + 4);
  };
  auto storeRegs = [&](int buf) {
    #pragma unroll
    for (int i = 0; i < 4; i++) {
      int row = (tid >> 3) + i * 32;
      __bf16* d = &As[buf][row][ac4];
      d[0] = f2bf(arg[i].x); d[1] = f2bf(arg[i].y);
      d[2] = f2bf(arg[i].z); d[3] = f2bf(arg[i].w);
    }
    Bs[buf][bn8 + 0][bkk] = f2bf(brg[0].x);
    Bs[buf][bn8 + 1][bkk] = f2bf(brg[0].y);
    Bs[buf][bn8 + 2][bkk] = f2bf(brg[0].z);
    Bs[buf][bn8 + 3][bkk] = f2bf(brg[0].w);
    Bs[buf][bn8 + 4][bkk] = f2bf(brg[1].x);
    Bs[buf][bn8 + 5][bkk] = f2bf(brg[1].y);
    Bs[buf][bn8 + 6][bkk] = f2bf(brg[1].z);
    Bs[buf][bn8 + 7][bkk] = f2bf(brg[1].w);
  };

  floatx8 c00 = {}, c01 = {}, c10 = {}, c11 = {};

  loadRegs(0);
  storeRegs(0);
  __syncthreads();

  const int nk = K / BK;
  for (int t = 0; t < nk; t++) {
    const int cur = t & 1;
    if (t + 1 < nk) loadRegs((t + 1) * BK);   // global loads in flight over WMMA
    if (t + 2 < nk) {                         // cache-prime tile t+2 (global_prefetch_b8)
      __builtin_prefetch(A + (long)(blockM + (tid >> 3)) * K + (t + 2) * BK + ac4, 0, 1);
      __builtin_prefetch(W + baseN + (long)((t + 2) * BK + bkk) * sK + bn8, 0, 1);
    }

    bf16x16 a0, a1, b0, b1;
    const __bf16* r0 = &As[cur][m0 + am][0];
    const __bf16* r1 = &As[cur][m0 + 16 + am][0];
    #pragma unroll
    for (int i = 0; i < 8; i++) {
      a0[i] = r0[akb + i];  a0[8 + i] = r0[akb + 16 + i];
      a1[i] = r1[akb + i];  a1[8 + i] = r1[akb + 16 + i];
    }
    const __bf16* p0 = &Bs[cur][n0 + bn][0];
    const __bf16* p1 = &Bs[cur][n0 + 16 + bn][0];
    #pragma unroll
    for (int i = 0; i < 16; i++) {
      b0[i] = p0[bk0 + i + (i & 8)];
      b1[i] = p1[bk0 + i + (i & 8)];
    }

    c00 = __builtin_amdgcn_wmma_f32_16x16x32_bf16(false, a0, false, b0, (short)0, c00, false, false);
    c01 = __builtin_amdgcn_wmma_f32_16x16x32_bf16(false, a0, false, b1, (short)0, c01, false, false);
    c10 = __builtin_amdgcn_wmma_f32_16x16x32_bf16(false, a1, false, b0, (short)0, c10, false, false);
    c11 = __builtin_amdgcn_wmma_f32_16x16x32_bf16(false, a1, false, b1, (short)0, c11, false, false);

    if (t + 1 < nk) storeRegs(cur ^ 1);
    __syncthreads();
  }

  // epilogue: C layout VGPR r -> M=r (lanes 0-15) / M=8+r (lanes 16-31)
  const int cm8 = (lane >> 4) * 8;
  const int cn  = lane & 15;
  auto store_tile = [&](const floatx8& cc, int rb, int cb) {
    #pragma unroll
    for (int r = 0; r < 8; r++) {
      int gm = rb + cm8 + r;
      int gn = cb + cn;
      float val = cc[r] + bias[gn];
      if (flags & FLAG_GELU)
        val = 0.5f * val * (1.f + erff(val * 0.70710678118654752f));
      long idx;
      if (flags & FLAG_QKV) {
        int bb = gm >> 11;            // / S_
        int ss = gm & (S_ - 1);
        int hh = gn >> 6, hd = gn & 63;
        idx = ((long)(bb * H_ + hh) * S_ + ss) * HD_ + hd;
      } else {
        idx = (long)gm * N + gn;
      }
      if (flags & FLAG_RES) C[idx] += val; else C[idx] = val;
    }
  };
  store_tile(c00, blockM + m0,      blockN + n0);
  store_tile(c01, blockM + m0,      blockN + n0 + 16);
  store_tile(c10, blockM + m0 + 16, blockN + n0);
  store_tile(c11, blockM + m0 + 16, blockN + n0 + 16);
}

// -------------------------- WMMA flash attention --------------------------
// grid (B*H, S/128); 8 waves; each wave owns 16 query rows. K/V tiles of 32 keys.
#define ATK 32
__global__ __launch_bounds__(256) void attn_wmma(const float* __restrict__ Q,
                                                 const float* __restrict__ Kx,
                                                 const float* __restrict__ V,
                                                 float* __restrict__ av)
{
  const int bh = blockIdx.x;        // b*H + h
  const int b  = bh >> 4;
  const int hh = bh & (H_ - 1);
  const int tid = threadIdx.x, lane = tid & 31, wave = tid >> 5;
  const int qrow0 = blockIdx.y * 128 + wave * 16;

  __shared__ __bf16 Ks[ATK][HD_ + 8];     // [key][feat]   32*72*2 = 4608 B
  __shared__ __bf16 Vt[HD_][ATK + 8];     // [feat][key]   64*40*2 = 5120 B
  __shared__ __bf16 Ps[8][16][ATK + 8];   // per-wave P    8*16*40*2 = 10240 B

  const int am = lane & 15, akb = (lane >> 4) * 8;
  const int bn = lane & 15, bk0 = (lane >> 4) * 8;

  // Q fragments (scaled by 1/sqrt(HD)), held in registers for the whole kernel
  bf16x16 aq0, aq1;
  {
    const float* qr = Q + ((long)bh * S_ + qrow0 + am) * HD_;
    #pragma unroll
    for (int i = 0; i < 8; i++) {
      aq0[i]     = f2bf(qr[akb + i]      * 0.125f);
      aq0[8 + i] = f2bf(qr[akb + 16 + i] * 0.125f);
      aq1[i]     = f2bf(qr[32 + akb + i]      * 0.125f);
      aq1[8 + i] = f2bf(qr[32 + akb + 16 + i] * 0.125f);
    }
  }

  float m8[8], l8[8];
  floatx8 o0 = {}, o1 = {}, o2 = {}, o3 = {};
  #pragma unroll
  for (int r = 0; r < 8; r++) { m8[r] = -3.0e38f; l8[r] = 0.f; }

  for (int t0 = 0; t0 < S_; t0 += ATK) {
    __syncthreads();
    {   // stage K tile [key][feat] and V tile transposed [feat][key], bf16
      int r = tid >> 3, c8 = (tid & 7) * 8;
      const float* kp = Kx + ((long)bh * S_ + t0 + r) * HD_ + c8;
      const float* vp = V  + ((long)bh * S_ + t0 + r) * HD_ + c8;
      float4 k0 = *(const float4*)kp, k1 = *(const float4*)(kp + 4);
      float4 v0 = *(const float4*)vp, v1 = *(const float4*)(vp + 4);
      if (t0 + ATK < S_) {            // prime next tile (global_prefetch_b8)
        __builtin_prefetch(kp + (long)ATK * HD_, 0, 1);
        __builtin_prefetch(vp + (long)ATK * HD_, 0, 1);
      }
      __bf16* d = &Ks[r][c8];
      d[0] = f2bf(k0.x); d[1] = f2bf(k0.y); d[2] = f2bf(k0.z); d[3] = f2bf(k0.w);
      d[4] = f2bf(k1.x); d[5] = f2bf(k1.y); d[6] = f2bf(k1.z); d[7] = f2bf(k1.w);
      Vt[c8 + 0][r] = f2bf(v0.x); Vt[c8 + 1][r] = f2bf(v0.y);
      Vt[c8 + 2][r] = f2bf(v0.z); Vt[c8 + 3][r] = f2bf(v0.w);
      Vt[c8 + 4][r] = f2bf(v1.x); Vt[c8 + 5][r] = f2bf(v1.y);
      Vt[c8 + 6][r] = f2bf(v1.z); Vt[c8 + 7][r] = f2bf(v1.w);
    }
    __syncthreads();

    // ---- scores: two 16-key subtiles, S = Q(16x64) . K^T ----
    floatx8 s0 = {}, s1 = {};
    {
      bf16x16 bk_lo, bk_hi;
      const __bf16* kr = &Ks[bn][0];
      #pragma unroll
      for (int i = 0; i < 16; i++) { bk_lo[i] = kr[bk0 + i + (i & 8)]; bk_hi[i] = kr[32 + bk0 + i + (i & 8)]; }
      s0 = __builtin_amdgcn_wmma_f32_16x16x32_bf16(false, aq0, false, bk_lo, (short)0, s0, false, false);
      s0 = __builtin_amdgcn_wmma_f32_16x16x32_bf16(false, aq1, false, bk_hi, (short)0, s0, false, false);
      const __bf16* kr2 = &Ks[16 + bn][0];
      #pragma unroll
      for (int i = 0; i < 16; i++) { bk_lo[i] = kr2[bk0 + i + (i & 8)]; bk_hi[i] = kr2[32 + bk0 + i + (i & 8)]; }
      s1 = __builtin_amdgcn_wmma_f32_16x16x32_bf16(false, aq0, false, bk_lo, (short)0, s1, false, false);
      s1 = __builtin_amdgcn_wmma_f32_16x16x32_bf16(false, aq1, false, bk_hi, (short)0, s1, false, false);
    }

    // ---- online softmax (rows live in 16-lane groups of C layout) ----
    float alpha[8];
    #pragma unroll
    for (int r = 0; r < 8; r++) {
      float mx = fmaxf(s0[r], s1[r]);
      mx = fmaxf(mx, __shfl_xor(mx, 1));
      mx = fmaxf(mx, __shfl_xor(mx, 2));
      mx = fmaxf(mx, __shfl_xor(mx, 4));
      mx = fmaxf(mx, __shfl_xor(mx, 8));
      float nm = fmaxf(m8[r], mx);
      alpha[r] = __expf(m8[r] - nm);
      m8[r] = nm;
      float p0 = __expf(s0[r] - nm);
      float p1 = __expf(s1[r] - nm);
      s0[r] = p0; s1[r] = p1;
      float ls = p0 + p1;
      ls += __shfl_xor(ls, 1);
      ls += __shfl_xor(ls, 2);
      ls += __shfl_xor(ls, 4);
      ls += __shfl_xor(ls, 8);
      l8[r] = l8[r] * alpha[r] + ls;
    }

    // ---- P: C layout -> LDS row-major [row][key], reload as A fragment ----
    #pragma unroll
    for (int r = 0; r < 8; r++) {
      int row = r + ((lane >> 4) * 8);
      Ps[wave][row][lane & 15]        = f2bf(s0[r]);
      Ps[wave][row][16 + (lane & 15)] = f2bf(s1[r]);
    }
    asm volatile("" ::: "memory");   // keep LDS store->load ordered (same wave, in-order DS)
    bf16x16 ap;
    {
      const __bf16* pr = &Ps[wave][am][0];
      #pragma unroll
      for (int i = 0; i < 8; i++) { ap[i] = pr[akb + i]; ap[8 + i] = pr[akb + 16 + i]; }
    }

    // ---- O = O*alpha + P(16x32) . V(32x64) : 4 feature-group WMMAs ----
    #pragma unroll
    for (int r = 0; r < 8; r++) {
      o0[r] *= alpha[r]; o1[r] *= alpha[r]; o2[r] *= alpha[r]; o3[r] *= alpha[r];
    }
    bf16x16 bv0, bv1, bv2, bv3;
    {
      const __bf16* v0p = &Vt[ 0 + bn][0];
      const __bf16* v1p = &Vt[16 + bn][0];
      const __bf16* v2p = &Vt[32 + bn][0];
      const __bf16* v3p = &Vt[48 + bn][0];
      #pragma unroll
      for (int i = 0; i < 16; i++) {
        bv0[i] = v0p[bk0 + i + (i & 8)]; bv1[i] = v1p[bk0 + i + (i & 8)];
        bv2[i] = v2p[bk0 + i + (i & 8)]; bv3[i] = v3p[bk0 + i + (i & 8)];
      }
    }
    o0 = __builtin_amdgcn_wmma_f32_16x16x32_bf16(false, ap, false, bv0, (short)0, o0, false, false);
    o1 = __builtin_amdgcn_wmma_f32_16x16x32_bf16(false, ap, false, bv1, (short)0, o1, false, false);
    o2 = __builtin_amdgcn_wmma_f32_16x16x32_bf16(false, ap, false, bv2, (short)0, o2, false, false);
    o3 = __builtin_amdgcn_wmma_f32_16x16x32_bf16(false, ap, false, bv3, (short)0, o3, false, false);
  }

  // ---- finalize: O /= l, write av in [B,S,D] so Wo GEMM is plain ----
  float inv[8];
  #pragma unroll
  for (int r = 0; r < 8; r++) inv[r] = 1.f / l8[r];
  const int cm8 = (lane >> 4) * 8;
  #pragma unroll
  for (int r = 0; r < 8; r++) {
    int row = qrow0 + cm8 + r;
    float* out = av + ((long)(b * S_ + row)) * D_ + hh * HD_ + (lane & 15);
    out[0]  = o0[r] * inv[r];
    out[16] = o1[r] * inv[r];
    out[32] = o2[r] * inv[r];
    out[48] = o3[r] * inv[r];
  }
}

// -------------------------- output copy (async-LDS bounce when available) --------------------------
__global__ __launch_bounds__(256) void copy_kernel(const float* __restrict__ src,
                                                   float* __restrict__ dst, int n) {
#if USE_ASYNC_COPY
  __shared__ float buf[256 * 4];
  long base = (long)blockIdx.x * 1024 + (long)threadIdx.x * 4;
  if (base + 4 <= (long)n) {
    g_i32x4* g = (g_i32x4*)(src + base);
    l_i32x4* l = (l_i32x4*)(&buf[threadIdx.x * 4]);
    __builtin_amdgcn_global_load_async_to_lds_b128(g, l, 0, 0);
    __builtin_amdgcn_s_wait_asynccnt(0);
    __builtin_amdgcn_global_store_async_from_lds_b128((g_i32x4*)(dst + base), l, 0, 0);
    __builtin_amdgcn_s_wait_asynccnt(0);
  } else {
    for (long i = base; i < (long)n && i < base + 4; i++) dst[i] = src[i];
  }
#else
  int i = blockIdx.x * 256 + threadIdx.x;
  if (i < n) dst[i] = src[i];
#endif
}

// -------------------------- launcher --------------------------
extern "C" void kernel_launch(void* const* d_in, const int* in_sizes, int n_in,
                              void* d_out, int out_size, void* d_ws, size_t ws_size,
                              hipStream_t stream) {
  const int*   ids   = (const int*)d_in[0];
  const float* tok   = (const float*)d_in[1];
  const float* pos   = (const float*)d_in[2];
  const float* emb_g = (const float*)d_in[3];
  const float* emb_b = (const float*)d_in[4];
  const float* ln1_g = (const float*)d_in[5];
  const float* ln1_b = (const float*)d_in[6];
  const float* ln2_g = (const float*)d_in[7];
  const float* ln2_b = (const float*)d_in[8];
  const float* Wq    = (const float*)d_in[9];
  const float* bq    = (const float*)d_in[10];
  const float* Wk    = (const float*)d_in[11];
  const float* bk    = (const float*)d_in[12];
  const float* Wv    = (const float*)d_in[13];
  const float* bv    = (const float*)d_in[14];
  const float* Wo    = (const float*)d_in[15];
  const float* bo    = (const float*)d_in[16];
  const float* W1    = (const float*)d_in[17];
  const float* b1    = (const float*)d_in[18];
  const float* W2    = (const float*)d_in[19];
  const float* b2    = (const float*)d_in[20];

  char* w = (char*)d_ws;
  const size_t XB = (size_t)M_ * D_ * sizeof(float);   // 16 MB per [B,S,D] buffer
  float* x  = (float*)(w);            // residual stream
  float* h  = (float*)(w + XB);       // LN output
  float* q  = (float*)(w + 2 * XB);   // [B,H,S,HD]
  float* kk = (float*)(w + 3 * XB);
  float* v  = (float*)(w + 4 * XB);
  float* av = (float*)(w + 5 * XB);   // [B,S,D]
  float* f  = (float*)(w + 6 * XB);   // [B,S,F] 64 MB

  embed_kernel<<<M_, 256, 0, stream>>>(ids, tok, pos, x);
  ln_kernel<<<M_, 256, 0, stream>>>(x, emb_g, emb_b, x);

  dim3 gD(D_ / BN, M_ / BM);     // N = 1024
  dim3 gF(F_ / BN, M_ / BM);     // N = 4096
  dim3 gA(B_ * H_, S_ / 128);

  for (int l = 0; l < L_; l++) {
    ln_kernel<<<M_, 256, 0, stream>>>(x, ln1_g + l * D_, ln1_b + l * D_, h);

    long wqkv = (long)l * H_ * D_ * HD_;
    gemm_bf16_wmma<<<gD, 256, 0, stream>>>(h, D_, Wq + wqkv, (long)D_ * HD_, HD_, 6,
                                           bq + l * H_ * HD_, q,  D_, FLAG_QKV);
    gemm_bf16_wmma<<<gD, 256, 0, stream>>>(h, D_, Wk + wqkv, (long)D_ * HD_, HD_, 6,
                                           bk + l * H_ * HD_, kk, D_, FLAG_QKV);
    gemm_bf16_wmma<<<gD, 256, 0, stream>>>(h, D_, Wv + wqkv, (long)D_ * HD_, HD_, 6,
                                           bv + l * H_ * HD_, v,  D_, FLAG_QKV);

    attn_wmma<<<gA, 256, 0, stream>>>(q, kk, v, av);

    gemm_bf16_wmma<<<gD, 256, 0, stream>>>(av, D_, Wo + (long)l * D_ * D_, 1, D_, 0,
                                           bo + l * D_, x, D_, FLAG_RES);

    ln_kernel<<<M_, 256, 0, stream>>>(x, ln2_g + l * D_, ln2_b + l * D_, h);

    gemm_bf16_wmma<<<gF, 256, 0, stream>>>(h, D_, W1 + (long)l * D_ * F_, 1, F_, 0,
                                           b1 + l * F_, f, F_, FLAG_GELU);
    gemm_bf16_wmma<<<gD, 256, 0, stream>>>(f, F_, W2 + (long)l * F_ * D_, 1, D_, 0,
                                           b2 + l * D_, x, D_, FLAG_RES);
  }

  copy_kernel<<<(out_size + 1023) / 1024, 256, 0, stream>>>(x, (float*)d_out, out_size);
}
